// RegimeConditionedMoE_86363202388379
// MI455X (gfx1250) — matrix operand order repaired and checked by reading, not compiled, not run on hardware
//
#include <hip/hip_runtime.h>
#include <cstdint>
#include <cstddef>

typedef __attribute__((ext_vector_type(16))) __bf16 v16bf;
typedef __attribute__((ext_vector_type(8)))  float  v8f;

namespace {

constexpr int BATCH = 2048;
constexpr int DIM   = 512;
constexpr int HID   = 2048;
constexpr int OUTD  = 512;
constexpr int NE    = 8;
constexpr int NR    = 4;
constexpr int FEAT  = DIM + NR;     // 516
constexpr int KPAD  = 544;          // FEAT padded to multiple of 32
constexpr int MPAD  = BATCH + NR * 128;  // 2560 padded-sample rows
constexpr int MROWS = MPAD * NE;         // 20480 (sample,expert) rows
constexpr float LNEPS = 1e-5f;

constexpr int BM = 128, BN = 128, BK = 32, LDA = 40;  // LDS row stride (bf16) for bank spread

union Pack8  { uint4 q; __bf16 h[8]; };
union FragBF { v16bf v; uint4 q[2]; };

// ---------------- A-tile loaders (produce 8 packed bf16 = 16B) ----------------
struct ALoadBf16 {
  const __bf16* A; int lda;
  __device__ __forceinline__ uint4 operator()(int r, int k) const {
    return *(const uint4*)(A + (size_t)r * lda + k);
  }
};

struct ALoadF32 {
  const float* A; int lda;
  __device__ __forceinline__ uint4 operator()(int r, int k) const {
    const float* p = A + (size_t)r * lda + k;
    float4 f0 = *(const float4*)p;
    float4 f1 = *(const float4*)(p + 4);
    Pack8 u;
    u.h[0] = (__bf16)f0.x; u.h[1] = (__bf16)f0.y; u.h[2] = (__bf16)f0.z; u.h[3] = (__bf16)f0.w;
    u.h[4] = (__bf16)f1.x; u.h[5] = (__bf16)f1.y; u.h[6] = (__bf16)f1.z; u.h[7] = (__bf16)f1.w;
    return u.q;
  }
};

// LayerNorm(+gain/bias)+ReLU applied while loading A (K = normalized axis)
struct ALoadLN {
  const __bf16* A; int lda;
  const float* mean; const float* rstd;   // per global row
  const float* g; const float* bb;        // per K column
  __device__ __forceinline__ uint4 operator()(int r, int k) const {
    Pack8 in; in.q = *(const uint4*)(A + (size_t)r * lda + k);
    float m = mean[r], rs = rstd[r];
    Pack8 out;
#pragma unroll
    for (int j = 0; j < 8; ++j) {
      float f = ((float)in.h[j] - m) * rs * g[k + j] + bb[k + j];
      out.h[j] = (__bf16)(f > 0.f ? f : 0.f);
    }
    return out.q;
  }
};

// ---------------- epilogues ----------------
struct EpReluBf16 {
  __bf16* C; int ldc; const float* bias;
  __device__ __forceinline__ void operator()(int r, int c, float v) const {
    float f = v + bias[c];
    C[(size_t)r * ldc + c] = (__bf16)(f > 0.f ? f : 0.f);
  }
};
struct EpBf16 {
  __bf16* C; int ldc; const float* bias;
  __device__ __forceinline__ void operator()(int r, int c, float v) const {
    C[(size_t)r * ldc + c] = (__bf16)(v + bias[c]);
  }
};
struct EpF32Res {
  float* C; int ldc; const float* bias; const float* res; int ldres;
  __device__ __forceinline__ void operator()(int r, int c, float v) const {
    C[(size_t)r * ldc + c] = v + bias[c] + res[(size_t)r * ldres + c];
  }
};
struct EpF32Add {
  float* C; int ldc; const float* bias;
  __device__ __forceinline__ void operator()(int r, int c, float v) const {
    size_t idx = (size_t)r * ldc + c;
    C[idx] = C[idx] + v + bias[c];
  }
};

// ------- WMMA tile GEMM core: 256 threads, 128x128 block tile, 32x64 wave tile, K step 32 -------
template <class AL, class EP>
__device__ __forceinline__ void gemm_tile(const AL& aload, const EP& ep,
                                          const __bf16* __restrict__ Bt, int ldb,
                                          int m0, int n0, int K) {
  __shared__ __attribute__((aligned(16))) __bf16 As[BM * LDA];
  __shared__ __attribute__((aligned(16))) __bf16 Bs[BN * LDA];
  const int t = threadIdx.x;
  const int wave = t >> 5, lane = t & 31;
  const int wm = wave & 3, wn = wave >> 2;     // 4 x 2 wave grid -> 32x64 per wave
  const int lrow = lane & 15, lhi = lane >> 4;

  const v8f vzero = {0.f, 0.f, 0.f, 0.f, 0.f, 0.f, 0.f, 0.f};
  v8f acc[2][4];
#pragma unroll
  for (int i = 0; i < 2; ++i)
#pragma unroll
    for (int j = 0; j < 4; ++j) acc[i][j] = vzero;

  for (int k0 = 0; k0 < K; k0 += BK) {
    // cooperative global->LDS loads (16B per thread per row-chunk)
#pragma unroll
    for (int p = 0; p < 2; ++p) {
      int r = p * 64 + (t >> 2);
      int kc = (t & 3) * 8;
      *(uint4*)(As + r * LDA + kc) = aload(m0 + r, k0 + kc);
      *(uint4*)(Bs + r * LDA + kc) = *(const uint4*)(Bt + (size_t)(n0 + r) * ldb + k0 + kc);
    }
    __syncthreads();
    FragBF a[2], b[4];
#pragma unroll
    for (int fm = 0; fm < 2; ++fm) {
      int ar = wm * 32 + fm * 16 + lrow;
      int kh = lhi * 8;   // A layout: K chunks {kh..kh+7} U {kh+16..kh+23}
      a[fm].q[0] = *(const uint4*)(As + ar * LDA + kh);
      a[fm].q[1] = *(const uint4*)(As + ar * LDA + kh + 16);
    }
#pragma unroll
    for (int fn = 0; fn < 4; ++fn) {
      int br = wn * 64 + fn * 16 + lrow;
      int kb = lhi * 8;   // B layout: K chunks {kb..kb+7} U {kb+16..kb+23}
      b[fn].q[0] = *(const uint4*)(Bs + br * LDA + kb);
      b[fn].q[1] = *(const uint4*)(Bs + br * LDA + kb + 16);
    }
#pragma unroll
    for (int fm = 0; fm < 2; ++fm)
#pragma unroll
      for (int fn = 0; fn < 4; ++fn)
        acc[fm][fn] = __builtin_amdgcn_wmma_f32_16x16x32_bf16(
            false, a[fm].v, false, b[fn].v, (short)0, acc[fm][fn], false, false);
    __syncthreads();
  }
  // C layout: lane l, vgpr j -> row lhi*8+j, col l%16
#pragma unroll
  for (int fm = 0; fm < 2; ++fm)
#pragma unroll
    for (int fn = 0; fn < 4; ++fn)
#pragma unroll
      for (int j = 0; j < 8; ++j) {
        int r = m0 + wm * 32 + fm * 16 + lhi * 8 + j;
        int c = n0 + wn * 64 + fn * 16 + lrow;
        ep(r, c, acc[fm][fn][j]);
      }
}

// ---------------- GEMM wrapper kernels ----------------
__global__ __launch_bounds__(256) void k_gemm_router(const __bf16* Xg, const __bf16* rw1t,
                                                     const float* rb1, __bf16* rh,
                                                     const int* segoff) {
  int r = blockIdx.z;
  int m0 = segoff[r] + blockIdx.y * BM;
  if (m0 >= segoff[r + 1]) return;
  ALoadBf16 al{Xg, KPAD};
  EpReluBf16 ep{rh, HID, rb1 + r * HID};
  gemm_tile(al, ep, rw1t + (size_t)r * HID * KPAD, KPAD, m0, blockIdx.x * BN, KPAD);
}

__global__ __launch_bounds__(256) void k_gemm_exp1(const __bf16* Xg, const __bf16* ew1t,
                                                   const float* eb1, __bf16* hbuf,
                                                   const int* segoff) {
  int z = blockIdx.z, r = z >> 3, e = z & 7;
  int m0 = segoff[r] + blockIdx.y * BM;
  if (m0 >= segoff[r + 1]) return;
  ALoadBf16 al{Xg, KPAD};
  EpBf16 ep{hbuf + (size_t)e * MPAD * HID, HID, eb1 + (size_t)z * HID};
  gemm_tile(al, ep, ew1t + (size_t)z * HID * DIM, DIM, m0, blockIdx.x * BN, DIM);
}

__global__ __launch_bounds__(256) void k_gemm_exp2(const __bf16* hbuf, const __bf16* ew2t,
                                                   const float* eb2, const float* lnm,
                                                   const float* lnr, const float* lng,
                                                   const float* lnb, const float* xf,
                                                   float* eo, const int* segoff) {
  int z = blockIdx.z, r = z >> 3, e = z & 7;
  int m0 = segoff[r] + blockIdx.y * BM;
  if (m0 >= segoff[r + 1]) return;
  ALoadLN al{hbuf + (size_t)e * MPAD * HID, HID,
             lnm + (size_t)e * MPAD, lnr + (size_t)e * MPAD,
             lng + (size_t)z * HID, lnb + (size_t)z * HID};
  EpF32Res ep{eo + (size_t)e * OUTD, NE * OUTD, eb2 + (size_t)z * OUTD, xf, DIM};
  gemm_tile(al, ep, ew2t + (size_t)z * OUTD * HID, HID, m0, blockIdx.x * BN, HID);
}

__global__ __launch_bounds__(256) void k_gemm_qkv(const float* eo, const __bf16* ipwb,
                                                  const float* ipb, __bf16* qkvb) {
  ALoadF32 al{eo, OUTD};
  EpBf16 ep{qkvb, 3 * OUTD, ipb};
  gemm_tile(al, ep, ipwb, OUTD, blockIdx.y * BM, blockIdx.x * BN, OUTD);
}

__global__ __launch_bounds__(256) void k_gemm_oproj(const __bf16* avb, const __bf16* opwb,
                                                    const float* opb, float* eo) {
  ALoadBf16 al{avb, OUTD};
  EpF32Add ep{eo, OUTD, opb};
  gemm_tile(al, ep, opwb, OUTD, blockIdx.y * BM, blockIdx.x * BN, OUTD);
}

// ---------------- setup / small kernels ----------------
__global__ void k_count(const int* regime, int* cnt) {
  int b = blockIdx.x * 256 + threadIdx.x;
  if (b < BATCH) atomicAdd(&cnt[regime[b]], 1);
}

__global__ void k_offsets(const int* cnt, int* segoff) {
  if (blockIdx.x == 0 && threadIdx.x == 0) {
    int o = 0;
    for (int r = 0; r < NR; ++r) { segoff[r] = o; o += ((cnt[r] + 127) >> 7) << 7; }
    segoff[NR] = o;
  }
}

__global__ void k_scatter(const int* regime, const int* segoff, int* fill,
                          int* rowmap, int* rowOf) {
  int b = blockIdx.x * 256 + threadIdx.x;
  if (b >= BATCH) return;
  int r = regime[b];
  int pos = segoff[r] + atomicAdd(&fill[r], 1);
  rowmap[pos] = b;
  rowOf[b] = pos;
}

__global__ __launch_bounds__(256) void k_gather(const int* rowmap, const float* x,
                                                const float* rf, __bf16* Xg, float* xf) {
  int i = blockIdx.x;
  int b = rowmap[i];
  if (b < 0) return;  // padding row stays zero (memset)
  int t = threadIdx.x;
  float v0 = x[(size_t)b * DIM + t];
  float v1 = x[(size_t)b * DIM + t + 256];
  Xg[(size_t)i * KPAD + t] = (__bf16)v0;
  Xg[(size_t)i * KPAD + t + 256] = (__bf16)v1;
  xf[(size_t)i * DIM + t] = v0;
  xf[(size_t)i * DIM + t + 256] = v1;
  if (t < NR) Xg[(size_t)i * KPAD + DIM + t] = (__bf16)rf[(size_t)b * NR + t];
}

// batched tiled transpose + fp32->bf16: dst[b][c][r] = src[b][r][c]
__global__ void k_transpose(const float* src, __bf16* dst, int rows, int cols, int dstride) {
  __shared__ float tile[32][33];
  int b = blockIdx.z;
  int r0 = blockIdx.x * 32, c0 = blockIdx.y * 32;
#pragma unroll
  for (int yy = 0; yy < 4; ++yy) {
    int r = r0 + threadIdx.y + yy * 8;
    int c = c0 + threadIdx.x;
    tile[threadIdx.y + yy * 8][threadIdx.x] =
        (r < rows && c < cols) ? src[((size_t)b * rows + r) * cols + c] : 0.f;
  }
  __syncthreads();
#pragma unroll
  for (int yy = 0; yy < 4; ++yy) {
    int c = c0 + threadIdx.y + yy * 8;   // output row (= src col)
    int r = r0 + threadIdx.x;            // output col (= src row)
    if (c < cols && r < rows)
      dst[((size_t)b * cols + c) * dstride + r] = (__bf16)tile[threadIdx.x][threadIdx.y + yy * 8];
  }
}

__global__ void k_convert(const float* src, __bf16* dst, int n) {
  int i = blockIdx.x * 256 + threadIdx.x;
  if (i < n) dst[i] = (__bf16)src[i];
}

__global__ __launch_bounds__(256) void k_logits(const __bf16* rh, const float* rw2,
                                                const float* rb2, const int* segoff,
                                                float* logits) {
  int i = blockIdx.x;
  int r = 0;
#pragma unroll
  for (int j = 1; j < NR; ++j) if (i >= segoff[j]) r = j;
  int t = threadIdx.x;
  float p[NE];
#pragma unroll
  for (int e = 0; e < NE; ++e) p[e] = 0.f;
  const __bf16* a = rh + (size_t)i * HID;
  const float* w = rw2 + (size_t)r * HID * NE;
  for (int k = t; k < HID; k += 256) {
    float f = (float)a[k];
    const float* wr = w + (size_t)k * NE;
#pragma unroll
    for (int e = 0; e < NE; ++e) p[e] += f * wr[e];
  }
  __shared__ float red[256 * NE];
#pragma unroll
  for (int e = 0; e < NE; ++e) red[t * NE + e] = p[e];
  __syncthreads();
  for (int st = 128; st > 0; st >>= 1) {
    if (t < st)
#pragma unroll
      for (int e = 0; e < NE; ++e) red[t * NE + e] += red[(t + st) * NE + e];
    __syncthreads();
  }
  if (t < NE) logits[(size_t)i * NE + t] = red[t] + rb2[r * NE + t];
}

__global__ void k_gate(const float* logits, float* tkw, int* tki) {
  int i = blockIdx.x * 256 + threadIdx.x;
  if (i >= MPAD) return;
  float l[NE];
  float mx = -1e30f;
#pragma unroll
  for (int e = 0; e < NE; ++e) { l[e] = logits[(size_t)i * NE + e]; mx = fmaxf(mx, l[e]); }
#pragma unroll
  for (int e = 0; e < NE; ++e) l[e] = __expf(l[e] - mx);
  int i0 = 0;
#pragma unroll
  for (int e = 1; e < NE; ++e) if (l[e] > l[i0]) i0 = e;
  int i1 = (i0 == 0) ? 1 : 0;
#pragma unroll
  for (int e = 0; e < NE; ++e) if (e != i0 && l[e] > l[i1]) i1 = e;
  float z = l[i0] + l[i1];
  tkw[i * 2] = l[i0] / z; tkw[i * 2 + 1] = l[i1] / z;
  tki[i * 2] = i0; tki[i * 2 + 1] = i1;
}

__global__ __launch_bounds__(256) void k_lnstats(const __bf16* hbuf, float* lnm, float* lnr) {
  int i = blockIdx.x, e = blockIdx.y;
  const __bf16* p = hbuf + ((size_t)e * MPAD + i) * HID;
  int t = threadIdx.x;
  float s = 0.f, s2 = 0.f;
  for (int k = t; k < HID; k += 256) { float f = (float)p[k]; s += f; s2 += f * f; }
  __shared__ float r1[256], r2[256];
  r1[t] = s; r2[t] = s2;
  __syncthreads();
  for (int st = 128; st > 0; st >>= 1) {
    if (t < st) { r1[t] += r1[t + st]; r2[t] += r2[t + st]; }
    __syncthreads();
  }
  if (t == 0) {
    float mean = r1[0] / (float)HID;
    float var = r2[0] / (float)HID - mean * mean;
    var = var < 0.f ? 0.f : var;
    lnm[(size_t)e * MPAD + i] = mean;
    lnr[(size_t)e * MPAD + i] = rsqrtf(var + LNEPS);
  }
}

__global__ __launch_bounds__(256) void k_attn(const __bf16* qkvb, __bf16* avb) {
  int i = blockIdx.x;
  __shared__ __attribute__((aligned(16))) __bf16 s[NE * 3 * OUTD];  // 8 x 1536
  const __bf16* src = qkvb + (size_t)i * NE * 3 * OUTD;
  for (int t = threadIdx.x; t < NE * 3 * OUTD / 8; t += 256)
    ((uint4*)s)[t] = ((const uint4*)src)[t];
  __syncthreads();
  int t = threadIdx.x;
  if (t < 64) {
    int hd = t >> 3, eq = t & 7;
    const __bf16* q = s + eq * 1536 + hd * 64;
    float sc[NE];
    float mx = -1e30f;
#pragma unroll
    for (int ek = 0; ek < NE; ++ek) {
      const __bf16* kk = s + ek * 1536 + 512 + hd * 64;
      float d = 0.f;
#pragma unroll
      for (int j = 0; j < 64; ++j) d += (float)q[j] * (float)kk[j];
      d *= 0.125f;  // 1/sqrt(64)
      sc[ek] = d; mx = fmaxf(mx, d);
    }
    float sum = 0.f;
#pragma unroll
    for (int ek = 0; ek < NE; ++ek) { sc[ek] = __expf(sc[ek] - mx); sum += sc[ek]; }
    float inv = 1.f / sum;
    __bf16* dst = avb + ((size_t)i * NE + eq) * OUTD + hd * 64;
#pragma unroll
    for (int j = 0; j < 64; ++j) {
      float a = 0.f;
#pragma unroll
      for (int ek = 0; ek < NE; ++ek) a += sc[ek] * (float)s[ek * 1536 + 1024 + hd * 64 + j];
      dst[j] = (__bf16)(a * inv);
    }
  }
}

__global__ __launch_bounds__(256) void k_lnfinal(float* eo, const float* g, const float* b) {
  int row = blockIdx.x;
  float* p = eo + (size_t)row * OUTD;
  int t = threadIdx.x;
  float v0 = p[t], v1 = p[t + 256];
  __shared__ float red[256];
  red[t] = v0 + v1;
  __syncthreads();
  for (int st = 128; st > 0; st >>= 1) { if (t < st) red[t] += red[t + st]; __syncthreads(); }
  float mean = red[0] / (float)OUTD;
  __syncthreads();
  float d0 = v0 - mean, d1 = v1 - mean;
  red[t] = d0 * d0 + d1 * d1;
  __syncthreads();
  for (int st = 128; st > 0; st >>= 1) { if (t < st) red[t] += red[t + st]; __syncthreads(); }
  float rs = rsqrtf(red[0] / (float)OUTD + LNEPS);
  p[t] = d0 * rs * g[t] + b[t];
  p[t + 256] = d1 * rs * g[t + 256] + b[t + 256];
}

__global__ __launch_bounds__(256) void k_combine(const float* eo, const float* tkw,
                                                 const int* tki, const int* rowOf, float* out) {
  int b = blockIdx.x;
  int i = rowOf[b];
  float w0 = tkw[i * 2], w1 = tkw[i * 2 + 1];
  int e0 = tki[i * 2], e1 = tki[i * 2 + 1];
  const float* p0 = eo + ((size_t)i * NE + e0) * OUTD;
  const float* p1 = eo + ((size_t)i * NE + e1) * OUTD;
  int t = threadIdx.x;
  out[(size_t)b * OUTD + t] = w0 * p0[t] + w1 * p1[t];
  out[(size_t)b * OUTD + t + 256] = w0 * p0[t + 256] + w1 * p1[t + 256];
}

}  // namespace

extern "C" void kernel_launch(void* const* d_in, const int* in_sizes, int n_in,
                              void* d_out, int out_size, void* d_ws, size_t ws_size,
                              hipStream_t stream) {
  const float* x    = (const float*)d_in[0];
  const int*   reg  = (const int*)d_in[1];
  const float* rf   = (const float*)d_in[2];
  const float* ew1  = (const float*)d_in[3];
  const float* eb1  = (const float*)d_in[4];
  const float* lng  = (const float*)d_in[5];
  const float* lnb  = (const float*)d_in[6];
  const float* ew2  = (const float*)d_in[7];
  const float* eb2  = (const float*)d_in[8];
  const float* rw1  = (const float*)d_in[9];
  const float* rb1  = (const float*)d_in[10];
  const float* rw2  = (const float*)d_in[11];
  const float* rb2  = (const float*)d_in[12];
  const float* ipw  = (const float*)d_in[13];
  const float* ipb  = (const float*)d_in[14];
  const float* opw  = (const float*)d_in[15];
  const float* opb  = (const float*)d_in[16];
  const float* ang  = (const float*)d_in[17];
  const float* anb  = (const float*)d_in[18];
  float* out = (float*)d_out;

  // ---- workspace layout (~375 MB), 256B aligned slabs ----
  uint8_t* w = (uint8_t*)d_ws;
  size_t off = 0;
  auto take = [&](size_t bytes) -> void* {
    void* p = w + off;
    off = (off + bytes + 255) & ~(size_t)255;
    return p;
  };
  int* cnt      = (int*)take(NR * 4);
  int* fill     = (int*)take(NR * 4);
  int* segoff   = (int*)take((NR + 1) * 4);
  int* rowmap   = (int*)take(MPAD * 4);
  int* rowOf    = (int*)take(BATCH * 4);
  __bf16* Xg    = (__bf16*)take((size_t)MPAD * KPAD * 2);
  float* xf     = (float*)take((size_t)MPAD * DIM * 4);
  __bf16* rw1t  = (__bf16*)take((size_t)NR * HID * KPAD * 2);
  __bf16* ew1t  = (__bf16*)take((size_t)NR * NE * HID * DIM * 2);
  __bf16* ew2t  = (__bf16*)take((size_t)NR * NE * OUTD * HID * 2);
  __bf16* ipwb  = (__bf16*)take((size_t)3 * OUTD * DIM * 2);
  __bf16* opwb  = (__bf16*)take((size_t)OUTD * OUTD * 2);
  __bf16* rh    = (__bf16*)take((size_t)MPAD * HID * 2);
  float* logits = (float*)take((size_t)MPAD * NE * 4);
  float* tkw    = (float*)take((size_t)MPAD * 2 * 4);
  int*   tki    = (int*)take((size_t)MPAD * 2 * 4);
  __bf16* hbuf  = (__bf16*)take((size_t)NE * MPAD * HID * 2);
  float* lnm    = (float*)take((size_t)NE * MPAD * 4);
  float* lnr    = (float*)take((size_t)NE * MPAD * 4);
  float* eo     = (float*)take((size_t)MROWS * OUTD * 4);
  __bf16* qkvb  = (__bf16*)take((size_t)MROWS * 3 * OUTD * 2);
  __bf16* avb   = (__bf16*)take((size_t)MROWS * OUTD * 2);

  // ---- init (graph-capture-safe async memsets) ----
  hipMemsetAsync(cnt, 0, NR * 4, stream);
  hipMemsetAsync(fill, 0, NR * 4, stream);
  hipMemsetAsync(rowmap, 0xFF, MPAD * 4, stream);
  hipMemsetAsync(Xg, 0, (size_t)MPAD * KPAD * 2, stream);
  hipMemsetAsync(xf, 0, (size_t)MPAD * DIM * 4, stream);
  hipMemsetAsync(rw1t, 0, (size_t)NR * HID * KPAD * 2, stream);
  hipMemsetAsync(eo, 0, (size_t)MROWS * OUTD * 4, stream);

  // ---- regime grouping ----
  k_count<<<(BATCH + 255) / 256, 256, 0, stream>>>(reg, cnt);
  k_offsets<<<1, 1, 0, stream>>>(cnt, segoff);
  k_scatter<<<(BATCH + 255) / 256, 256, 0, stream>>>(reg, segoff, fill, rowmap, rowOf);
  k_gather<<<MPAD, 256, 0, stream>>>(rowmap, x, rf, Xg, xf);

  // ---- weight prep: transpose + bf16 ----
  {
    dim3 blk(32, 8);
    k_transpose<<<dim3(17, HID / 32, NR), blk, 0, stream>>>(rw1, rw1t, FEAT, HID, KPAD);
    k_transpose<<<dim3(DIM / 32, HID / 32, NR * NE), blk, 0, stream>>>(ew1, ew1t, DIM, HID, DIM);
    k_transpose<<<dim3(HID / 32, OUTD / 32, NR * NE), blk, 0, stream>>>(ew2, ew2t, HID, OUTD, HID);
  }
  k_convert<<<(3 * OUTD * DIM + 255) / 256, 256, 0, stream>>>(ipw, ipwb, 3 * OUTD * DIM);
  k_convert<<<(OUTD * OUTD + 255) / 256, 256, 0, stream>>>(opw, opwb, OUTD * OUTD);

  // ---- router + gating ----
  k_gemm_router<<<dim3(HID / BN, 16, NR), 256, 0, stream>>>(Xg, rw1t, rb1, rh, segoff);
  k_logits<<<MPAD, 256, 0, stream>>>(rh, rw2, rb2, segoff, logits);
  k_gate<<<(MPAD + 255) / 256, 256, 0, stream>>>(logits, tkw, tki);

  // ---- experts ----
  k_gemm_exp1<<<dim3(HID / BN, 16, NR * NE), 256, 0, stream>>>(Xg, ew1t, eb1, hbuf, segoff);
  k_lnstats<<<dim3(MPAD, NE), 256, 0, stream>>>(hbuf, lnm, lnr);
  k_gemm_exp2<<<dim3(OUTD / BN, 16, NR * NE), 256, 0, stream>>>(hbuf, ew2t, eb2, lnm, lnr,
                                                                lng, lnb, xf, eo, segoff);

  // ---- attention over expert axis ----
  k_gemm_qkv<<<dim3(3 * OUTD / BN, MROWS / BM), 256, 0, stream>>>(eo, ipwb, ipb, qkvb);
  k_attn<<<MPAD, 256, 0, stream>>>(qkvb, avb);
  k_gemm_oproj<<<dim3(OUTD / BN, MROWS / BM), 256, 0, stream>>>(avb, opwb, opb, eo);
  k_lnfinal<<<MROWS, 256, 0, stream>>>(eo, ang, anb);

  // ---- top-k combine ----
  k_combine<<<BATCH, 256, 0, stream>>>(eo, tkw, tki, rowOf, out);
}
